// HRRSimpleSelfAttention_58634893525656
// MI455X (gfx1250) — matrix operand order repaired and checked by reading, not compiled, not run on hardware
//
#include <hip/hip_runtime.h>
#include <hip/hip_bf16.h>

// ---------------------------------------------------------------------------
// HRR self-attention (FFT-binding) pipeline for MI455X (gfx1250, wave32).
//   B=4, S=4096, D=2048, H=16, hd=128  (reference shapes)
//
// Everything heavy is WMMA bf16 (f32 accum), staged through LDS with
// gfx1250 async global->LDS DMA:
//   qh  = x @ Wq.T                      (GEMM, bf16 out)
//   kvh = x @ Wkv.T                     (GEMM, bf16 out)
//   qf  = qh  @ F.T  ; kvf = kvh @ F.T  (DFT-as-GEMM, F = [cos;sin] 256x128)
//   normalize qf, kvf per (b,s,h) vector (complex L2)
//   kvf = cumsum_S(kvf)                 (causal, serial scan)
//   Zcat = qf (*) kvf                   (complex product, packed [re|im] bf16)
//   vh  = Zcat @ Wi.T                   (iDFT-as-GEMM, Wi = [cos/-sin]/128, bf16 out)
//   out = vh @ Wo.T                     (GEMM, f32 out -> d_out)
// ---------------------------------------------------------------------------

typedef __attribute__((ext_vector_type(16))) __bf16 v16bf;
typedef __attribute__((ext_vector_type(8)))  __bf16 v8bf;
typedef __attribute__((ext_vector_type(8)))  float  v8f;

#define PI_F 3.14159265358979323846f

// Use gfx1250 async global->LDS DMA (ASYNCcnt-tracked) for GEMM tile staging.
#define USE_ASYNC_LDS 1

// 16-byte async copy: global (64-bit vgpr addr) -> LDS (32-bit offset vgpr).
// VDST holds the LDS byte offset from LDS_BASE; low 32 bits of a generic
// pointer to a __shared__ object are exactly that offset.
__device__ __forceinline__ void async_copy_b128(void* lds_dst, const void* gsrc) {
#if USE_ASYNC_LDS
    unsigned lds_off = (unsigned)(unsigned long long)lds_dst;
    unsigned long long gaddr = (unsigned long long)gsrc;
    asm volatile("global_load_async_to_lds_b128 %0, %1, off"
                 :: "v"(lds_off), "v"(gaddr) : "memory");
#else
    *(v8bf*)lds_dst = *(const v8bf*)gsrc;
#endif
}

__device__ __forceinline__ void async_wait_all() {
#if USE_ASYNC_LDS
    asm volatile("s_wait_asynccnt 0" ::: "memory");
#endif
}

// ------------------------------ f32 -> bf16 --------------------------------
__global__ void cvt_f32_bf16_kernel(const float* __restrict__ in,
                                    __bf16* __restrict__ out, size_t n) {
    size_t i = (size_t)blockIdx.x * blockDim.x + threadIdx.x;
    size_t stride = (size_t)gridDim.x * blockDim.x;
    for (; i < n; i += stride) out[i] = (__bf16)in[i];
}

// --------------------------- twiddle generation ----------------------------
// F (256x128): rows 0..127 = cos(-2*pi*k*n/128), rows 128..255 = sin(-2*pi*k*n/128)
__global__ void gen_dft_kernel(__bf16* __restrict__ F) {
    int i = blockIdx.x * blockDim.x + threadIdx.x;     // 0 .. 256*128-1
    if (i >= 256 * 128) return;
    int r = i >> 7, n = i & 127;
    int k = r & 127;
    float ang = -2.0f * PI_F * (float)((k * n) & 127) / 128.0f;
    F[i] = (__bf16)((r < 128) ? cosf(ang) : sinf(ang));
}
// Wi (128x256): Wi[t][k]      =  cos(2*pi*t*k/128)/128        (k < 128)
//               Wi[t][128+k'] = -sin(2*pi*t*k'/128)/128
__global__ void gen_idft_kernel(__bf16* __restrict__ Wi) {
    int i = blockIdx.x * blockDim.x + threadIdx.x;     // 0 .. 128*256-1
    if (i >= 128 * 256) return;
    int t = i >> 8, k = i & 255;
    int kk = k & 127;
    float ang = 2.0f * PI_F * (float)((t * kk) & 127) / 128.0f;
    float v = (k < 128) ? cosf(ang) : -sinf(ang);
    Wi[i] = (__bf16)(v * (1.0f / 128.0f));
}

// ------------------------------ WMMA GEMM ----------------------------------
// C[M,N] = A[M,K] (bf16, row-major) * Bw[N,K]^T (bf16, row-major); OutT store.
// Block: 256 threads = 8 waves (4 M-waves x 2 N-waves); block tile 128x128.
// Wave tile: 32 (M) x 64 (N) = 2x4 WMMA 16x16x32 accumulators.
// Per K-step (BK=32): 128x32 A and B tiles staged in LDS via async DMA,
// double-buffered; fragments read with ds_load_b128.
// Requires M%128==0, N%128==0, K%32==0.
#define BK    32
#define LDP   40   // padded LDS row stride in elements (80B: 16B-aligned, bank-friendly)

template <typename OutT>
__global__ __launch_bounds__(256)
void gemm_bf16_wmma_kernel(const __bf16* __restrict__ A,
                           const __bf16* __restrict__ Bw,
                           OutT* __restrict__ C,
                           int M, int N, int K) {
    __shared__ __bf16 sA[2][128 * LDP];
    __shared__ __bf16 sB[2][128 * LDP];

    const int tid   = threadIdx.x;
    const int lane  = tid & 31;
    const int wave  = tid >> 5;
    const int row16 = lane & 15;   // row within 16x16 tile / C column index
    const int half  = lane >> 4;   // selects K sub-range / C row group

    const int wm = wave & 3;       // 0..3 -> M offset within block tile
    const int wn = wave >> 2;      // 0..1 -> N offset within block tile
    const int mblk = blockIdx.y * 128;
    const int nblk = blockIdx.x * 128;

    // Tile copy: 128 rows x 32 cols bf16 = 512 x 16B chunks; 256 threads x 2.
    auto stage_tiles = [&](int buf, int k0) {
#pragma unroll
        for (int i = 0; i < 2; ++i) {
            const int c   = tid + 256 * i;   // chunk id 0..511
            const int row = c >> 2;
            const int seg = (c & 3) * 8;     // element offset within row
            async_copy_b128(&sA[buf][row * LDP + seg],
                            A + (size_t)(mblk + row) * K + k0 + seg);
            async_copy_b128(&sB[buf][row * LDP + seg],
                            Bw + (size_t)(nblk + row) * K + k0 + seg);
        }
    };

    v8f acc[2][4] = {};

    stage_tiles(0, 0);
    const int ksteps = K / BK;
    for (int kt = 0; kt < ksteps; ++kt) {
        async_wait_all();
        __syncthreads();               // all waves' DMA into buf now visible
        const int buf = kt & 1;
        if (kt + 1 < ksteps) stage_tiles(buf ^ 1, (kt + 1) * BK);

        // A fragments (16x32 per tile):
        // lanes 0-15 (half=0): elems 0..7 = K[0..7],  8..15 = K[16..23]
        // lanes16-31 (half=1): elems 0..7 = K[8..15], 8..15 = K[24..31]
        union { v16bf v; v8bf h[2]; } afr[2];
#pragma unroll
        for (int i = 0; i < 2; ++i) {
            const __bf16* ap = &sA[buf][(wm * 32 + i * 16 + row16) * LDP + half * 8];
            afr[i].h[0] = *(const v8bf*)(ap);
            afr[i].h[1] = *(const v8bf*)(ap + 16);
        }
        // B fragments (32x16 per tile): column n = row16; same K order as A:
        // half=0: K[0..7],K[16..23]; half=1: K[8..15],K[24..31].
        union { v16bf v; v8bf h[2]; } bfr[4];
#pragma unroll
        for (int j = 0; j < 4; ++j) {
            const __bf16* bp = &sB[buf][(wn * 64 + j * 16 + row16) * LDP + half * 8];
            bfr[j].h[0] = *(const v8bf*)(bp);
            bfr[j].h[1] = *(const v8bf*)(bp + 16);
        }
#pragma unroll
        for (int i = 0; i < 2; ++i)
#pragma unroll
            for (int j = 0; j < 4; ++j)
                acc[i][j] = __builtin_amdgcn_wmma_f32_16x16x32_bf16(
                    /*neg_a=*/false, afr[i].v,
                    /*neg_b=*/false, bfr[j].v,
                    /*c_mod=*/(short)0, acc[i][j],
                    /*reuse_a=*/false, /*reuse_b=*/false);
    }

    // C layout: VGPR r -> row (r + 8*half), col = row16.
    const int m0 = mblk + wm * 32;
    const int n0 = nblk + wn * 64;
#pragma unroll
    for (int i = 0; i < 2; ++i) {
#pragma unroll
        for (int j = 0; j < 4; ++j) {
            const int col = n0 + j * 16 + row16;
#pragma unroll
            for (int r = 0; r < 8; ++r) {
                const int rowm = m0 + i * 16 + r + half * 8;
                C[(size_t)rowm * N + col] = (OutT)acc[i][j][r];
            }
        }
    }
}

// ------------------- per-vector complex L2 normalization -------------------
// Spectra packed per (b,s,h) row as [re(128) | im(128)] f32.
// One block of 128 threads per vector; gridDim.x = B*S*H.
__global__ __launch_bounds__(128)
void norm_kernel(float* __restrict__ QF) {
    __shared__ float red[128];
    const int t = threadIdx.x;
    const size_t v = blockIdx.x;
    float re = QF[v * 256 + t];
    float im = QF[v * 256 + 128 + t];
    red[t] = re * re + im * im;
    __syncthreads();
    for (int off = 64; off > 0; off >>= 1) {
        if (t < off) red[t] += red[t + off];
        __syncthreads();
    }
    const float inv = rsqrtf(red[0]);
    QF[v * 256 + t]       = re * inv;
    QF[v * 256 + 128 + t] = im * inv;
}

// --------------------- causal cumsum over sequence dim ---------------------
// 256 channels (re+im packed) per (b,h); one block of 256 threads per (b,h).
__global__ __launch_bounds__(256)
void cumsum_kernel(float* __restrict__ KF, int S, int H) {
    const int t  = threadIdx.x;            // channel 0..255
    const int bh = blockIdx.x;
    const int b  = bh / H;
    const int h  = bh % H;
    float a = 0.f;
    const size_t stride = (size_t)H * 256;
    size_t idx = ((size_t)b * S * H + h) * 256 + t;
    for (int s = 0; s < S; ++s, idx += stride) {
        if (s + 1 < S) __builtin_prefetch(&KF[idx + stride], 1, 1);
        a += KF[idx];
        KF[idx] = a;
    }
}

// ----------------- complex bin product -> packed bf16 [re|im] --------------
__global__ void cmul_kernel(const float* __restrict__ QF,
                            const float* __restrict__ KF,
                            __bf16* __restrict__ Zcat, size_t nvec) {
    size_t i = (size_t)blockIdx.x * blockDim.x + threadIdx.x;   // vec*128 + bin
    size_t stride = (size_t)gridDim.x * blockDim.x;
    const size_t total = nvec * 128;
    for (; i < total; i += stride) {
        const size_t v = i >> 7, t = i & 127;
        const size_t o = v * 256 + t;
        const float qr = QF[o], qi = QF[o + 128];
        const float kr = KF[o], ki = KF[o + 128];
        Zcat[o]       = (__bf16)(qr * kr - qi * ki);
        Zcat[o + 128] = (__bf16)(qr * ki + qi * kr);
    }
}

// ---------------------------------------------------------------------------
extern "C" void kernel_launch(void* const* d_in, const int* in_sizes, int n_in,
                              void* d_out, int out_size, void* d_ws, size_t ws_size,
                              hipStream_t stream) {
    (void)in_sizes; (void)n_in; (void)out_size; (void)ws_size;
    const int B = 4, S = 4096, D = 2048, H = 16;
    const int BS = B * S;                 // 16384
    const size_t ND = (size_t)BS * D;     // 33,554,432 elements
    const int nvec = BS * H;              // 262144 (b,s,h) vectors of length 128

    const float* x   = (const float*)d_in[0];
    const float* Wq  = (const float*)d_in[1];
    const float* Wkv = (const float*)d_in[2];
    const float* Wo  = (const float*)d_in[3];
    float* out = (float*)d_out;

    // --- workspace carve-up (256B aligned) ---
    char* ws = (char*)d_ws;
    size_t off = 0;
    auto carve = [&](size_t bytes) -> char* {
        char* p = ws + off;
        off = (off + bytes + 255) & ~(size_t)255;
        return p;
    };
    __bf16* xh   = (__bf16*)carve(ND * 2);
    __bf16* wqh  = (__bf16*)carve((size_t)D * D * 2);
    __bf16* wkvh = (__bf16*)carve((size_t)D * D * 2);
    __bf16* woh  = (__bf16*)carve((size_t)D * D * 2);
    __bf16* Ftw  = (__bf16*)carve((size_t)256 * 128 * 2);  // DFT twiddles
    __bf16* Wi   = (__bf16*)carve((size_t)128 * 256 * 2);  // iDFT twiddles
    __bf16* qh   = (__bf16*)carve(ND * 2);                 // q  (bf16)
    __bf16* kvh  = (__bf16*)carve(ND * 2);                 // kv (bf16)
    float*  qf   = (float*)carve((size_t)nvec * 256 * 4);  // q spectrum [re|im]
    float*  kvf  = (float*)carve((size_t)nvec * 256 * 4);  // kv spectrum [re|im]
    __bf16* zcat = (__bf16*)carve((size_t)nvec * 256 * 2); // bound spectrum (bf16)
    __bf16* vh   = (__bf16*)carve(ND * 2);                 // unbound values (bf16)

    // 1) conversions + twiddle tables
    cvt_f32_bf16_kernel<<<2048, 256, 0, stream>>>(x, xh, ND);
    cvt_f32_bf16_kernel<<<512, 256, 0, stream>>>(Wq, wqh, (size_t)D * D);
    cvt_f32_bf16_kernel<<<512, 256, 0, stream>>>(Wkv, wkvh, (size_t)D * D);
    cvt_f32_bf16_kernel<<<512, 256, 0, stream>>>(Wo, woh, (size_t)D * D);
    gen_dft_kernel<<<(256 * 128 + 255) / 256, 256, 0, stream>>>(Ftw);
    gen_idft_kernel<<<(128 * 256 + 255) / 256, 256, 0, stream>>>(Wi);

    // 2) projections: q = x@Wq.T, kv = x@Wkv.T   (bf16 out)
    dim3 projGrid(D / 128, BS / 128);     // (16, 128)
    gemm_bf16_wmma_kernel<__bf16><<<projGrid, 256, 0, stream>>>(xh, wqh,  qh,  BS, D, D);
    gemm_bf16_wmma_kernel<__bf16><<<projGrid, 256, 0, stream>>>(xh, wkvh, kvh, BS, D, D);

    // 3) DFT as GEMM: (nvec x 128) @ F(256x128)^T -> (nvec x 256) packed [re|im]
    dim3 dftGrid(256 / 128, nvec / 128);  // (2, 2048)
    gemm_bf16_wmma_kernel<float><<<dftGrid, 256, 0, stream>>>(qh,  Ftw, qf,  nvec, 256, 128);
    gemm_bf16_wmma_kernel<float><<<dftGrid, 256, 0, stream>>>(kvh, Ftw, kvf, nvec, 256, 128);

    // 4) complex L2 normalize
    norm_kernel<<<nvec, 128, 0, stream>>>(qf);
    norm_kernel<<<nvec, 128, 0, stream>>>(kvf);

    // 5) causal cumsum of kvf over S
    cumsum_kernel<<<B * H, 256, 0, stream>>>(kvf, S, H);

    // 6) complex bin product -> packed bf16
    cmul_kernel<<<2048, 256, 0, stream>>>(qf, kvf, zcat, (size_t)nvec);

    // 7) iDFT as GEMM: (nvec x 256) @ Wi(128x256)^T -> vh (nvec x 128) bf16
    dim3 idftGrid(128 / 128, nvec / 128); // (1, 2048)
    gemm_bf16_wmma_kernel<__bf16><<<idftGrid, 256, 0, stream>>>(zcat, Wi, vh, nvec, 128, 256);

    // 8) out = vh @ Wo.T  -> f32 d_out
    gemm_bf16_wmma_kernel<float><<<projGrid, 256, 0, stream>>>(vh, woh, out, BS, D, D);
}
